// NS_CT_TotalSimRetina_26448408609544
// MI455X (gfx1250) — matrix operand order repaired and checked by reading, not compiled, not run on hardware
//
#include <hip/hip_runtime.h>
#include <hip/hip_bf16.h>

typedef __attribute__((ext_vector_type(16))) __bf16 v16bf;
typedef __attribute__((ext_vector_type(8)))  float  v8f;

// ---------------------------------------------------------------------------
// Kernel 0: zero the global-barrier state each launch (graph-replay safe).
// ---------------------------------------------------------------------------
__global__ void init_bar_kernel(unsigned* bar) {
    if (threadIdx.x < 4) bar[threadIdx.x] = 0u;
}

// ---------------------------------------------------------------------------
// Kernel 1: spat = frames(M=n_frames, K=n_pix) @ filters(N=n_cells, K)^T
// bf16 WMMA, f32 accumulate. One wave per 16x16 C tile, K unrolled 2x (64/iter).
// A layout (16-bit 16x32): lane L row m=L%16; elems 0..7 -> K=8*(L/16)+e,
// elems 8..15 -> K=16+8*(L/16)+(e-8).
// B layout (16-bit 32x16): lane L col n=L%16; same K mapping as A.
// ---------------------------------------------------------------------------
__device__ __forceinline__ void pack8_bf16(v16bf& v, int base, float4 lo, float4 hi) {
    v[base + 0] = (__bf16)lo.x; v[base + 1] = (__bf16)lo.y;
    v[base + 2] = (__bf16)lo.z; v[base + 3] = (__bf16)lo.w;
    v[base + 4] = (__bf16)hi.x; v[base + 5] = (__bf16)hi.y;
    v[base + 6] = (__bf16)hi.z; v[base + 7] = (__bf16)hi.w;
}

__device__ __forceinline__ v8f wmma_chunk32(const float* arow, const float* brow,
                                            int k0, v8f acc) {
    const float4* ap0 = (const float4*)(arow + k0);
    const float4* ap1 = (const float4*)(arow + k0 + 16);
    float4 a0 = ap0[0], a1 = ap0[1];   // K = base .. base+7
    float4 a2 = ap1[0], a3 = ap1[1];   // K = base+16 .. base+23
    const float4* bp = (const float4*)(brow + k0);
    const float4* bq = (const float4*)(brow + k0 + 16);
    float4 b0 = bp[0], b1 = bp[1], b2 = bq[0], b3 = bq[1];

    v16bf a, b;
    pack8_bf16(a, 0, a0, a1);
    pack8_bf16(a, 8, a2, a3);
    pack8_bf16(b, 0, b0, b1);
    pack8_bf16(b, 8, b2, b3);

    return __builtin_amdgcn_wmma_f32_16x16x32_bf16(
        /*neg_a=*/false, a, /*neg_b=*/false, b,
        /*c_mod=*/(short)0, acc, /*reuse_a=*/false, /*reuse_b=*/false);
}

__global__ void __launch_bounds__(128) gemm_spat_kernel(
    const float* __restrict__ frames,   // (M, K)
    const float* __restrict__ filt,     // (N, K)
    float* __restrict__ spat,           // (M, N)
    int M, int N, int K)
{
    const int lane = threadIdx.x & 31;
    const int wave = threadIdx.x >> 5;
    const int m0 = blockIdx.x * 16;
    const int n0 = blockIdx.y * 64 + wave * 16;
    const int hh = lane >> 4;    // 0 or 1 (lane half)
    const int lr = lane & 15;

    int am = m0 + lr; if (am >= M) am = M - 1;         // clamp padded rows
    int bn = n0 + lr; if (bn >= N) bn = N - 1;
    const float* arow = frames + (size_t)am * K + hh * 8;   // K base for A elems 0..7
    const float* brow = filt   + (size_t)bn * K + hh * 8;   // K base for B elems 0..7

    v8f acc = {};
    int k0 = 0;
    // main loop: 64 K-values (two WMMAs) per iteration
    for (; k0 + 64 <= K; k0 += 64) {
        __builtin_prefetch(arow + k0 + 512, 0, 0);
        __builtin_prefetch(brow + k0 + 512, 0, 0);
        acc = wmma_chunk32(arow, brow, k0,      acc);
        acc = wmma_chunk32(arow, brow, k0 + 32, acc);
    }
    for (; k0 < K; k0 += 32)
        acc = wmma_chunk32(arow, brow, k0, acc);

    // C layout: elem r of lane L -> m = r + 8*(L/16), n = L%16
    const int cn = n0 + lr;
    if (cn < N) {
        float* cbase = spat + (size_t)(m0 + hh * 8) * N + cn;
        if (m0 + 16 <= M) {
            // full tile: branchless strided stores
            #pragma unroll
            for (int r = 0; r < 8; ++r)
                cbase[(size_t)r * N] = acc[r];
        } else {
            #pragma unroll
            for (int r = 0; r < 8; ++r) {
                int cm = m0 + r + hh * 8;
                if (cm < M) spat[(size_t)cm * N + cn] = acc[r];
            }
        }
    }
}

// ---------------------------------------------------------------------------
// Kernel 2: stim[c][t] = bias[c] + sum_k fw[t][k] * spat[fsel[t][k]][c]
// ---------------------------------------------------------------------------
__global__ void stim_kernel(const float* __restrict__ spat,
                            const float* __restrict__ bias,
                            const float* __restrict__ fw,
                            const int* __restrict__ fsel,
                            float* __restrict__ stim,
                            int n_cells, int nbt)
{
    int i = blockIdx.x * blockDim.x + threadIdx.x;
    if (i >= n_cells * nbt) return;
    int c = i % n_cells;     // coalesced over c
    int t = i / n_cells;
    int s0 = fsel[t * 2 + 0];
    int s1 = fsel[t * 2 + 1];
    float v = bias[c]
            + fw[t * 2 + 0] * spat[(size_t)s0 * n_cells + c]
            + fw[t * 2 + 1] * spat[(size_t)s1 * n_cells + c];
    stim[(size_t)c * nbt + t] = v;
}

// ---------------------------------------------------------------------------
// Kernel 3: copy initial spikes into out[:, :n_init]
// ---------------------------------------------------------------------------
__global__ void copy_init_kernel(const float* __restrict__ init_spk,
                                 float* __restrict__ out,
                                 int n_cells, int n_init, int nbt)
{
    int i = blockIdx.x * blockDim.x + threadIdx.x;
    if (i >= n_cells * n_init) return;
    int c = i / n_init, j = i % n_init;
    out[(size_t)c * nbt + j] = init_spk[i];
}

// ---------------------------------------------------------------------------
// Kernel 4: persistent scan. 32 WGs x 512 threads (16 waves). Each WG keeps
// the FULL 512x100 f32 spike ring in LDS (200KB) + its own cells' coupling
// filters as bf16 (64KB) + feedback filters f32 + coupled_sel — 276KB, only
// feasible on CDNA5's 320KB LDS. One wave32 per owned cell; per-step cross-WG
// exchange is just 512 floats through a double-buffered global array guarded
// by an atomic generation barrier.
// ---------------------------------------------------------------------------
__global__ void __launch_bounds__(512, 1) scan_kernel(
    const float* __restrict__ init_spk,     // (n_cells, n_init)
    const float* __restrict__ ffilt_g,      // (n_cells, nb)
    const float* __restrict__ cfilt_g,      // (n_cells, max_c, nb)
    const int*   __restrict__ csel_g,       // (n_cells, max_c)
    const float* __restrict__ stim,         // (n_cells, nbt)
    float* __restrict__ out,                // (n_cells, nbt)
    float* __restrict__ spikes_buf,         // 2 * n_cells (double buffer)
    unsigned* __restrict__ bar,             // [0]=count, [1]=generation
    int n_cells, int nb, int max_c, int n_init, int nbt,
    int nwg, int cells_per_wg)
{
    extern __shared__ char smem[];
    float*  win = (float*)smem;                                        // n_cells*nb f32
    __bf16* cf  = (__bf16*)(smem + (size_t)n_cells * nb * 4);          // cpw*max_c*nb bf16
    float*  ff  = (float*)((char*)cf + (size_t)cells_per_wg * max_c * nb * 2);
    int*    cs  = (int*)((char*)ff + (size_t)cells_per_wg * nb * 4);

    const int tid  = threadIdx.x;
    const int lane = tid & 31;
    const int wid  = tid >> 5;                  // 0..cells_per_wg-1
    const int cell0 = blockIdx.x * cells_per_wg;
    const int cell  = cell0 + wid;

    // ---- load LDS state ----
    for (int i = tid; i < n_cells * nb; i += blockDim.x) {
        int c = i / nb, b = i % nb;
        win[i] = init_spk[(size_t)c * n_init + (n_init - nb) + b];
    }
    for (int i = tid; i < cells_per_wg * max_c * nb; i += blockDim.x)
        cf[i] = (__bf16)cfilt_g[(size_t)cell0 * max_c * nb + i];
    for (int i = tid; i < cells_per_wg * nb; i += blockDim.x)
        ff[i] = ffilt_g[(size_t)cell0 * nb + i];
    for (int i = tid; i < cells_per_wg * max_c; i += blockDim.x)
        cs[i] = csel_g[cell0 * max_c + i];
    __syncthreads();

    const int nsteps = nbt - n_init;
    const float*  stim_row = stim + (size_t)cell * nbt + (n_init - 1);
    const __bf16* mycf = cf + wid * max_c * nb;
    const float*  myff = ff + wid * nb;
    const int*    mycs = cs + wid * max_c;

    for (int i = 0; i < nsteps; ++i) {
        const int h = i % nb;                   // ring head (oldest slot)
        __builtin_prefetch(stim_row + i + 1, 0, 0);

        float sum = 0.f;
        // feedback: window[cell] . feedback_filter[cell]
        for (int b = lane; b < nb; b += 32) {
            int p = h + b; if (p >= nb) p -= nb;
            sum += win[cell * nb + p] * myff[b];
        }
        // coupling: sum_m window[sel[m]] . coupling_filter[cell][m]
        for (int m = 0; m < max_c; ++m) {
            const float*  wrow = win + mycs[m] * nb;
            const __bf16* crow = mycf + m * nb;
            for (int b = lane; b < nb; b += 32) {
                int p = h + b; if (p >= nb) p -= nb;
                sum += wrow[p] * (float)crow[b];
            }
        }
        // wave32 reduction
        #pragma unroll
        for (int off = 16; off > 0; off >>= 1)
            sum += __shfl_down(sum, off, 32);

        if (lane == 0) {
            float g = stim_row[i] + sum;
            float s = 1.0f / (1.0f + __expf(-g));
            spikes_buf[(i & 1) * n_cells + cell] = s;
            out[(size_t)cell * nbt + n_init + i] = s;
            __threadfence();                    // make spike device-visible
        }

        // ---- device-wide generation barrier ----
        __syncthreads();
        if (tid == 0) {
            volatile unsigned* cnt = bar;
            volatile unsigned* gen = bar + 1;
            unsigned g0 = *gen;
            __threadfence();
            if (atomicAdd((unsigned*)cnt, 1u) == (unsigned)(nwg - 1)) {
                *cnt = 0u;
                __threadfence();
                atomicAdd((unsigned*)gen, 1u);
            } else {
                while (*gen == g0) { __builtin_amdgcn_s_sleep(2); }
            }
        }
        __syncthreads();

        // ---- fold all 512 new spikes into the local ring ----
        for (int c = tid; c < n_cells; c += blockDim.x)
            win[c * nb + h] = spikes_buf[(i & 1) * n_cells + c];
        __syncthreads();
    }
}

// ---------------------------------------------------------------------------
extern "C" void kernel_launch(void* const* d_in, const int* in_sizes, int n_in,
                              void* d_out, int out_size, void* d_ws, size_t ws_size,
                              hipStream_t stream) {
    const float* initial_spikes   = (const float*)d_in[0];
    const float* input_frames     = (const float*)d_in[1];
    const float* spat_filters     = (const float*)d_in[2];
    const float* feedback_filters = (const float*)d_in[3];
    const float* coupling_filters = (const float*)d_in[4];
    const float* bias             = (const float*)d_in[5];
    const float* forward_weights  = (const float*)d_in[6];
    const int*   coupled_sel      = (const int*)d_in[7];
    const int*   forward_sel      = (const int*)d_in[8];
    float* out = (float*)d_out;

    const int n_cells = in_sizes[5];                 // 512
    const int nb      = in_sizes[3] / n_cells;       // 100
    const int n_init  = in_sizes[0] / n_cells;       // 100
    const int n_pix   = in_sizes[2] / n_cells;       // 25600
    const int n_frames= in_sizes[1] / n_pix;         // 300
    const int nbt     = in_sizes[6] / 2;             // 1500
    const int max_c   = in_sizes[7] / n_cells;       // 20

    // workspace layout
    char* ws = (char*)d_ws;
    size_t off = 0;
    float* spat = (float*)(ws + off);
    off += ((size_t)n_frames * n_cells * 4 + 255) & ~(size_t)255;
    float* stim = (float*)(ws + off);
    off += ((size_t)n_cells * nbt * 4 + 255) & ~(size_t)255;
    float* spikes_buf = (float*)(ws + off);
    off += ((size_t)2 * n_cells * 4 + 255) & ~(size_t)255;
    unsigned* bar = (unsigned*)(ws + off);

    // 0) reset barrier state (deterministic per launch / graph replay)
    init_bar_kernel<<<1, 64, 0, stream>>>(bar);

    // 1) WMMA bf16 GEMM: spat(n_frames, n_cells)
    dim3 ggrid((n_frames + 15) / 16, (n_cells + 63) / 64);
    gemm_spat_kernel<<<ggrid, 128, 0, stream>>>(
        input_frames, spat_filters, spat, n_frames, n_cells, n_pix);

    // 2) stim = gathered-weighted spat + bias
    int nst = n_cells * nbt;
    stim_kernel<<<(nst + 255) / 256, 256, 0, stream>>>(
        spat, bias, forward_weights, forward_sel, stim, n_cells, nbt);

    // 3) out[:, :n_init] = initial_spikes
    int ncp = n_cells * n_init;
    copy_init_kernel<<<(ncp + 255) / 256, 256, 0, stream>>>(
        initial_spikes, out, n_cells, n_init, nbt);

    // 4) persistent scan: 32 WGs, one wave32 per cell, 276KB LDS each
    const int nwg = 32;
    const int cells_per_wg = n_cells / nwg;          // 16
    const int block = cells_per_wg * 32;             // 512 threads = 16 waves
    size_t lds_bytes = (size_t)n_cells * nb * 4                     // window ring
                     + (size_t)cells_per_wg * max_c * nb * 2        // coupling bf16
                     + (size_t)cells_per_wg * nb * 4                // feedback f32
                     + (size_t)cells_per_wg * max_c * 4;            // coupled_sel
    scan_kernel<<<nwg, block, lds_bytes, stream>>>(
        initial_spikes, feedback_filters, coupling_filters, coupled_sel,
        stim, out, spikes_buf, bar,
        n_cells, nb, max_c, n_init, nbt, nwg, cells_per_wg);
}
